// GATTransformer_68805376081951
// MI455X (gfx1250) — matrix operand order repaired and checked by reading, not compiled, not run on hardware
//
#include <hip/hip_runtime.h>
#include <hip/hip_fp16.h>

#define N_NODES 50000
#define N_EDGES 800000
#define DIM 128
#define HEADS 2
#define DEPTH 2
#define MLP 512
#define HD (HEADS * DIM)          // 256
#define NEG_SLOPE 0.2f
#define LN_EPS 1e-5f

// GEMM tiling
#define GWAVES 8                  // waves per block
#define GMT 2                     // 16-row M-tiles per wave (keeps acc in 64 VGPRs)
#define LSTR 72                   // LDS B row stride in halfs (64 + 8 pad, 16B aligned)

typedef __attribute__((ext_vector_type(16))) _Float16 v16h;
typedef __attribute__((ext_vector_type(8)))  _Float16 v8h;
typedef __attribute__((ext_vector_type(8)))  float    v8f;
typedef __attribute__((ext_vector_type(4)))  float    v4f;

// ---------------------------------------------------------------- utilities

__global__ void cvt_f32_to_f16_kernel(const float* __restrict__ src,
                                      _Float16* __restrict__ dst, int n) {
  int i = blockIdx.x * blockDim.x + threadIdx.x;
  int stride = gridDim.x * blockDim.x;
  for (; i < n; i += stride) dst[i] = (_Float16)src[i];
}

__global__ void fill_f32_kernel(float* __restrict__ p, float v, int n) {
  int i = blockIdx.x * blockDim.x + threadIdx.x;
  int stride = gridDim.x * blockDim.x;
  for (; i < n; i += stride) p[i] = v;
}

__device__ __forceinline__ void atomicMaxFloat(float* addr, float v) {
  // classic two-sided trick: int-max for non-negative, uint-min for negative
  if (v >= 0.0f) atomicMax((int*)addr, __float_as_int(v));
  else           atomicMin((unsigned int*)addr, __float_as_uint(v));
}

__device__ __forceinline__ float gelu_tanh(float x) {
  const float k0 = 0.7978845608028654f;  // sqrt(2/pi)
  const float k1 = 0.044715f;
  float x3 = x * x * x;
  return 0.5f * x * (1.0f + tanhf(k0 * (x + k1 * x3)));
}

// ------------------------------------------------------------- LayerNorm
// one wave32 per node row of 128; writes f16 activations for WMMA GEMMs

__global__ void layernorm_f16_kernel(const float* __restrict__ x,
                                     const float* __restrict__ g,
                                     const float* __restrict__ b,
                                     _Float16* __restrict__ out) {
  int warp = (blockIdx.x * blockDim.x + threadIdx.x) >> 5;
  int lane = threadIdx.x & 31;
  if (warp >= N_NODES) return;
  const float* row = x + (size_t)warp * DIM;
  v4f v = *(const v4f*)(row + lane * 4);
  float s  = v[0] + v[1] + v[2] + v[3];
  float sq = v[0]*v[0] + v[1]*v[1] + v[2]*v[2] + v[3]*v[3];
  #pragma unroll
  for (int o = 16; o > 0; o >>= 1) {
    s  += __shfl_xor(s,  o, 32);
    sq += __shfl_xor(sq, o, 32);
  }
  float mean = s * (1.0f / DIM);
  float var  = sq * (1.0f / DIM) - mean * mean;
  float inv  = rsqrtf(var + LN_EPS);
  _Float16* orow = out + (size_t)warp * DIM;
  #pragma unroll
  for (int j = 0; j < 4; ++j) {
    int d = lane * 4 + j;
    orow[d] = (_Float16)((v[j] - mean) * inv * g[d] + b[d]);
  }
}

// ------------------------------------------------------------ WMMA GEMM
// C[M,N] = A[M,K](f16,row-major) @ B[K,N](f16,row-major) + bias.
// Weight-stationary: block stages the B panel [K x 64] in LDS once, then each
// wave computes GMT 16x64 output strips.  K is the outer loop so one set of B
// fragments (4 x v16h from ds_load) feeds GMT*4 = 8 WMMAs per K-step while
// A loads for the next M-tile overlap current WMMAs.
// Register budget (wave32): acc 2*4*8=64 + B frag 32 + A frag 8 + addr ~ 130
// VGPRs -> no spills, 2+ waves/SIMD.
// Epilogues: 0 = f32 store; 1 = f32 store + residual; 2 = tanh-GELU -> f16.
// ISA layouts (CDNA5 ISA §7.12.2):
//   A: lane L -> row M=L&15, halfs h=0..7 -> K=k0+(L>>4)*8+h ; h=8..15 -> +16
//   B: lane L -> col N=n0+(L&15), halfs h -> K as for A (strided LDS reads)
//   C/D: lane L -> col N=n0+(L&15), vgpr r -> row M=m0+r+8*(L>>4)

template <int EPI>
__global__ void __launch_bounds__(GWAVES * 32)
gemm_wmma_f16_kernel(const _Float16* __restrict__ A,
                     const _Float16* __restrict__ B,
                     const float* __restrict__ bias,
                     const float* __restrict__ res,
                     float* __restrict__ Cf32,
                     _Float16* __restrict__ Cf16,
                     int M, int N, int K) {
  extern __shared__ _Float16 ldsB[];          // [K][LSTR]
  int tid  = threadIdx.x;
  int wave = tid >> 5;
  int lane = tid & 31;
  int n0 = blockIdx.x << 6;                   // 64-column strip
  int mtiles = M >> 4;

  // ---- cooperative stage of B panel [K x 64] into LDS (16B chunks)
  for (int c = tid; c < K * 8; c += GWAVES * 32) {
    int row = c >> 3, cb = c & 7;
    *(v8h*)(ldsB + row * LSTR + cb * 8) =
        *(const v8h*)(B + (size_t)row * N + n0 + cb * 8);
  }
  __syncthreads();

  int lm = lane & 15;
  int g  = lane >> 4;
  int base = blockIdx.y * (GWAVES * GMT);

  int  mt_tile[GMT];
  bool mt_ok[GMT];
  #pragma unroll
  for (int q = 0; q < GMT; ++q) {
    int t = base + q * GWAVES + wave;         // wave-interleaved tiles
    mt_ok[q]   = (t < mtiles);                // wave-uniform
    mt_tile[q] = mt_ok[q] ? t : (mtiles - 1); // clamp: loads stay in-bounds
  }

  v8f acc[GMT][4];
  #pragma unroll
  for (int t = 0; t < 4; ++t) {
    float bv = bias[n0 + 16 * t + lm];
    #pragma unroll
    for (int q = 0; q < GMT; ++q)
      #pragma unroll
      for (int r = 0; r < 8; ++r) acc[q][t][r] = bv;
  }

  for (int k0 = 0; k0 < K; k0 += 32) {
    v16h bm[4];
    const _Float16* bcol = ldsB + k0 * LSTR + lm;
    #pragma unroll
    for (int t = 0; t < 4; ++t)
      #pragma unroll
      for (int s = 0; s < 16; ++s)
        bm[t][s] = bcol[(g * 8 + (s & 7) + ((s >> 3) << 4)) * LSTR + 16 * t];
    #pragma unroll
    for (int q = 0; q < GMT; ++q) {
      const _Float16* arow = A + (size_t)(mt_tile[q] * 16 + lm) * K + k0;
      union { v16h v; v8h h[2]; } a;
      a.h[0] = *(const v8h*)(arow + g * 8);
      a.h[1] = *(const v8h*)(arow + 16 + g * 8);
      #pragma unroll
      for (int t = 0; t < 4; ++t)
        acc[q][t] = __builtin_amdgcn_wmma_f32_16x16x32_f16(
            false, a.v, false, bm[t], (short)0, acc[q][t], false, false);
    }
  }

  #pragma unroll
  for (int q = 0; q < GMT; ++q) {
    if (!mt_ok[q]) continue;                  // wave-uniform guard
    int m0 = mt_tile[q] << 4;
    #pragma unroll
    for (int t = 0; t < 4; ++t) {
      #pragma unroll
      for (int r = 0; r < 8; ++r) {
        int m = m0 + r + 8 * g;
        size_t idx = (size_t)m * N + (n0 + 16 * t + lm);
        float v = acc[q][t][r];
        if (EPI == 1) v += res[idx];
        if (EPI == 2) Cf16[idx] = (_Float16)gelu_tanh(v);
        else          Cf32[idx] = v;
      }
    }
  }
}

// ---------------------------------------------------------- edge passes

// pass 1: score_eh = att_h . leaky_relu(xl[src] + xr[dst]); segment-max over dst
__global__ void edge_score_max_kernel(const float* __restrict__ xl,
                                      const float* __restrict__ xr,
                                      const float* __restrict__ att,
                                      const long long* __restrict__ src,
                                      const long long* __restrict__ dst,
                                      float* __restrict__ score,
                                      float* __restrict__ smax) {
  int e = (blockIdx.x * blockDim.x + threadIdx.x) >> 5;
  int lane = threadIdx.x & 31;
  if (e >= N_EDGES) return;
  int s = (int)src[e];
  int d = (int)dst[e];
  const float* pl = xl + (size_t)s * HD;
  const float* pr = xr + (size_t)d * HD;
  #pragma unroll
  for (int h = 0; h < HEADS; ++h) {
    float part = 0.0f;
    #pragma unroll
    for (int j = 0; j < DIM / 32; ++j) {
      int dd = lane + j * 32;
      float t = pl[h * DIM + dd] + pr[h * DIM + dd];
      t = (t > 0.0f) ? t : t * NEG_SLOPE;
      part += att[h * DIM + dd] * t;
    }
    #pragma unroll
    for (int o = 16; o > 0; o >>= 1) part += __shfl_xor(part, o, 32);
    if (lane == 0) {
      score[(size_t)e * HEADS + h] = part;
      atomicMaxFloat(&smax[(size_t)d * HEADS + h], part);
    }
  }
}

// pass 2: ex = exp(score - smax[dst]); segment-sum -> denom; score overwritten by ex
__global__ void edge_exp_sum_kernel(const long long* __restrict__ dst,
                                    const float* __restrict__ smax,
                                    float* __restrict__ score,
                                    float* __restrict__ denom) {
  int idx = blockIdx.x * blockDim.x + threadIdx.x;
  if (idx >= N_EDGES * HEADS) return;
  int e = idx >> 1;
  int h = idx & 1;
  int d = (int)dst[e];
  float ex = __expf(score[idx] - smax[(size_t)d * HEADS + h]);
  score[idx] = ex;
  atomicAdd(&denom[(size_t)d * HEADS + h], ex);
}

// pass 3: gat[dst] += alpha * xl[src]; one thread per (edge, head*dim) element
__global__ void edge_aggregate_kernel(const long long* __restrict__ src,
                                      const long long* __restrict__ dst,
                                      const float* __restrict__ xl,
                                      const float* __restrict__ ex,
                                      const float* __restrict__ denom,
                                      float* __restrict__ gat) {
  long long idx = (long long)blockIdx.x * blockDim.x + threadIdx.x;  // exact grid
  int e   = (int)(idx >> 8);      // / 256
  int rem = (int)(idx & 255);
  int h   = rem >> 7;
  int s = (int)src[e];
  int d = (int)dst[e];
  float a = ex[(size_t)e * HEADS + h] / (denom[(size_t)d * HEADS + h] + 1e-16f);
  atomicAdd(&gat[(size_t)d * HD + rem], xl[(size_t)s * HD + rem] * a);
}

// finalize: add gat_b, convert to f16 for the proj WMMA GEMM
__global__ void finalize_gat_kernel(const float* __restrict__ gat,
                                    const float* __restrict__ gat_b,
                                    _Float16* __restrict__ gat16) {
  int idx = blockIdx.x * blockDim.x + threadIdx.x;  // exact grid, N*HD
  gat16[idx] = (_Float16)(gat[idx] + gat_b[idx & (HD - 1)]);
}

// --------------------------------------------------------------- launch

extern "C" void kernel_launch(void* const* d_in, const int* in_sizes, int n_in,
                              void* d_out, int out_size, void* d_ws, size_t ws_size,
                              hipStream_t stream) {
  (void)in_sizes; (void)n_in; (void)out_size; (void)ws_size;

  const float*     x_in  = (const float*)d_in[0];
  const long long* eidx  = (const long long*)d_in[1];
  const float*     ln1_g = (const float*)d_in[2];
  const float*     ln1_b = (const float*)d_in[3];
  const float*     Wl    = (const float*)d_in[4];
  const float*     bl    = (const float*)d_in[5];
  const float*     Wr    = (const float*)d_in[6];
  const float*     br    = (const float*)d_in[7];
  const float*     att   = (const float*)d_in[8];
  const float*     gat_b = (const float*)d_in[9];
  const float*     projW = (const float*)d_in[10];
  const float*     projb = (const float*)d_in[11];
  const float*     ln2_g = (const float*)d_in[12];
  const float*     ln2_b = (const float*)d_in[13];
  const float*     W1    = (const float*)d_in[14];
  const float*     b1    = (const float*)d_in[15];
  const float*     W2    = (const float*)d_in[16];
  const float*     b2    = (const float*)d_in[17];

  const long long* srcI = eidx;
  const long long* dstI = eidx + N_EDGES;

  // -------- workspace layout (256B aligned slabs)
  char* base = (char*)d_ws;
  size_t off = 0;
  auto alloc = [&](size_t bytes) -> void* {
    off = (off + 255) & ~(size_t)255;
    void* p = base + off;
    off += bytes;
    return p;
  };
  float*    xbuf  = (float*)   alloc((size_t)N_NODES * DIM * 4);
  _Float16* h16   = (_Float16*)alloc((size_t)N_NODES * DIM * 2);
  float*    xl    = (float*)   alloc((size_t)N_NODES * HD  * 4);
  float*    xr    = (float*)   alloc((size_t)N_NODES * HD  * 4);
  float*    gat   = (float*)   alloc((size_t)N_NODES * HD  * 4);
  _Float16* gat16 = (_Float16*)alloc((size_t)N_NODES * HD  * 2);
  _Float16* hid16 = (_Float16*)alloc((size_t)N_NODES * MLP * 2);
  float*    smax  = (float*)   alloc((size_t)N_NODES * HEADS * 4);
  float*    denom = (float*)   alloc((size_t)N_NODES * HEADS * 4);
  float*    score = (float*)   alloc((size_t)N_EDGES * HEADS * 4);
  _Float16* wl16  = (_Float16*)alloc((size_t)DEPTH * DIM * HD  * 2);
  _Float16* wr16  = (_Float16*)alloc((size_t)DEPTH * DIM * HD  * 2);
  _Float16* pw16  = (_Float16*)alloc((size_t)DEPTH * HD  * DIM * 2);
  _Float16* w116  = (_Float16*)alloc((size_t)DEPTH * DIM * MLP * 2);
  _Float16* w216  = (_Float16*)alloc((size_t)DEPTH * MLP * DIM * 2);

  // -------- one-time weight conversion f32 -> f16
  auto cvt = [&](const float* s, _Float16* d, int n) {
    cvt_f32_to_f16_kernel<<<(n + 255) / 256, 256, 0, stream>>>(s, d, n);
  };
  cvt(Wl,    wl16, DEPTH * DIM * HD);
  cvt(Wr,    wr16, DEPTH * DIM * HD);
  cvt(projW, pw16, DEPTH * HD  * DIM);
  cvt(W1,    w116, DEPTH * DIM * MLP);
  cvt(W2,    w216, DEPTH * MLP * DIM);

  // running activation buffer (inputs must not be mutated)
  hipMemcpyAsync(xbuf, x_in, (size_t)N_NODES * DIM * 4,
                 hipMemcpyDeviceToDevice, stream);

  const int MTILES  = N_NODES / 16;                      // 3125, exact
  const int TILES_B = GWAVES * GMT;                      // 16 m-tiles per block
  const int gemm_gy = (MTILES + TILES_B - 1) / TILES_B;  // 196
  const dim3 gemm_block(GWAVES * 32);
  auto smem = [](int K) { return (size_t)K * LSTR * sizeof(_Float16); };

  for (int i = 0; i < DEPTH; ++i) {
    const _Float16* wl_i = wl16 + (size_t)i * DIM * HD;
    const _Float16* wr_i = wr16 + (size_t)i * DIM * HD;
    const _Float16* pw_i = pw16 + (size_t)i * HD  * DIM;
    const _Float16* w1_i = w116 + (size_t)i * DIM * MLP;
    const _Float16* w2_i = w216 + (size_t)i * MLP * DIM;

    // ---- PreNorm
    layernorm_f16_kernel<<<N_NODES * 32 / 256, 256, 0, stream>>>(
        xbuf, ln1_g + i * DIM, ln1_b + i * DIM, h16);

    // ---- xl = h @ Wl + bl ; xr = h @ Wr + br      [N, 256]
    gemm_wmma_f16_kernel<0>
        <<<dim3(HD / 64, gemm_gy), gemm_block, smem(DIM), stream>>>(
        h16, wl_i, bl + i * HD, nullptr, xl, nullptr, N_NODES, HD, DIM);
    gemm_wmma_f16_kernel<0>
        <<<dim3(HD / 64, gemm_gy), gemm_block, smem(DIM), stream>>>(
        h16, wr_i, br + i * HD, nullptr, xr, nullptr, N_NODES, HD, DIM);

    // ---- segment-softmax state init
    fill_f32_kernel<<<(N_NODES * HD + 255) / 256, 256, 0, stream>>>(
        gat, 0.0f, N_NODES * HD);
    fill_f32_kernel<<<(N_NODES * HEADS + 255) / 256, 256, 0, stream>>>(
        smax, -__builtin_huge_valf(), N_NODES * HEADS);
    fill_f32_kernel<<<(N_NODES * HEADS + 255) / 256, 256, 0, stream>>>(
        denom, 0.0f, N_NODES * HEADS);

    // ---- edge passes
    edge_score_max_kernel<<<N_EDGES * 32 / 256, 256, 0, stream>>>(
        xl, xr, att + i * HD, srcI, dstI, score, smax);
    edge_exp_sum_kernel<<<(N_EDGES * HEADS + 255) / 256, 256, 0, stream>>>(
        dstI, smax, score, denom);
    edge_aggregate_kernel<<<N_EDGES * HD / 256, 256, 0, stream>>>(
        srcI, dstI, xl, score, denom, gat);
    finalize_gat_kernel<<<N_NODES * HD / 256, 256, 0, stream>>>(
        gat, gat_b + i * HD, gat16);

    // ---- x = x + gat @ proj_W + proj_b            [N, 128]
    gemm_wmma_f16_kernel<1>
        <<<dim3(DIM / 64, gemm_gy), gemm_block, smem(HD), stream>>>(
        gat16, pw_i, projb + i * DIM, xbuf, xbuf, nullptr, N_NODES, DIM, HD);

    // ---- FFN: h = LN2(x); x = x + gelu(h@W1+b1)@W2 + b2
    layernorm_f16_kernel<<<N_NODES * 32 / 256, 256, 0, stream>>>(
        xbuf, ln2_g + i * DIM, ln2_b + i * DIM, h16);
    gemm_wmma_f16_kernel<2>
        <<<dim3(MLP / 64, gemm_gy), gemm_block, smem(DIM), stream>>>(
        h16, w1_i, b1 + i * MLP, nullptr, nullptr, hid16, N_NODES, MLP, DIM);
    gemm_wmma_f16_kernel<1>
        <<<dim3(DIM / 64, gemm_gy), gemm_block, smem(MLP), stream>>>(
        hid16, w2_i, b2 + i * DIM, xbuf, xbuf, nullptr, N_NODES, DIM, MLP);
  }

  hipMemcpyAsync(d_out, xbuf, (size_t)N_NODES * DIM * 4,
                 hipMemcpyDeviceToDevice, stream);
}
